// Attention_22926535426306
// MI455X (gfx1250) — matrix operand order repaired and checked by reading, not compiled, not run on hardware
//
#include <hip/hip_runtime.h>
#include <hip/hip_bf16.h>

// MI455X / gfx1250 flash-attention forward, all matmuls via v_wmma_f32_16x16x32_f16.
// Workspace layout (requires ws_size >= 48 MiB):
//   [ 0MB) featH  f16 [4096x1024]
//   [ 8MB) WqH    f16 [1024x1024]
//   [10MB) WkvH   f16 [2048x1024]
//   [14MB) WpH    f16 [1024x1024]
//   [16MB) Qh     f16 [H][N][D]   (pre-scaled by 1/sqrt(D))
//   [24MB) Kh     f16 [H][N][D]
//   [32MB) VtH    f16 [H][D][N]   (V transposed per head)
//   [40MB) Xh     f16 [N][C]      (attention output, pre output-projection)

typedef __attribute__((ext_vector_type(16))) _Float16 v16h;
typedef __attribute__((ext_vector_type(8)))  _Float16 v8h;
typedef __attribute__((ext_vector_type(8)))  float    v8f;
typedef __attribute__((ext_vector_type(4)))  float    v4f;

#define N_TOK      4096
#define C_DIM      1024
#define N_HEAD     16
#define D_HEAD     64
#define ATTN_SCALE 0.125f  // 64^-0.5

static __device__ __forceinline__ v8f wmma16(v16h a, v16h b, v8f c) {
  // (neg_a, A, neg_b, B, c_mod, C, reuse_a, reuse_b)
  return __builtin_amdgcn_wmma_f32_16x16x32_f16(false, a, false, b, (short)0, c,
                                                false, false);
}

// 16x32 f16 A fragment: per-lane halves [0..7] at p, [8..15] at p+16 elements.
// (Caller passes p = rowPtr + 8*(lane>>4); row = lane&15 per ISA A layout.)
// Works on both global and LDS pointers.
static __device__ __forceinline__ v16h ld_afrag(const _Float16* p) {
  v8h lo = *reinterpret_cast<const v8h*>(p);
  v8h hh = *reinterpret_cast<const v8h*>(p + 16);
  return __builtin_shufflevector(lo, hh, 0, 1, 2, 3, 4, 5, 6, 7,
                                         8, 9, 10, 11, 12, 13, 14, 15);
}

// ---------------------------------------------------------------------------
// f32 -> f16 convert (8 elements / thread, 16B vector stores)
// ---------------------------------------------------------------------------
__global__ __launch_bounds__(256) void cvt_f32_f16(const float* __restrict__ src,
                                                   _Float16* __restrict__ dst,
                                                   int n) {
  int i = (blockIdx.x * 256 + threadIdx.x) * 8;
  if (i >= n) return;
  v4f a = *reinterpret_cast<const v4f*>(src + i);
  v4f b = *reinterpret_cast<const v4f*>(src + i + 4);
  v8h o;
#pragma unroll
  for (int e = 0; e < 4; ++e) {
    o[e]     = (_Float16)a[e];
    o[e + 4] = (_Float16)b[e];
  }
  *reinterpret_cast<v8h*>(dst + i) = o;
}

// ---------------------------------------------------------------------------
// WMMA GEMM: out[m][n] = sum_k A[m][k] * W[n][k] + bias[n]
// Block: 256 thr (8 waves); wave tile 16 rows x 64 cols; block 128 x 64.
// Register double-buffered k-loop: fragments for k+32 are fetched before the
// WMMAs for k issue, so global loads stay in flight under the matrix ops.
// MODE 0: Q  -> Qh[h][n][d] * ATTN_SCALE        (f16)
// MODE 1: K  -> Kh[h][n][d]                     (f16)
// MODE 2: V  -> VtH[h][d][n]  (transposed)      (f16)
// MODE 3: final projection -> dstF[m][n]        (f32)
// ---------------------------------------------------------------------------
template <int MODE>
__global__ __launch_bounds__(256) void gemm_wmma(const _Float16* __restrict__ A,
                                                 const _Float16* __restrict__ W,
                                                 const float* __restrict__ bias,
                                                 _Float16* __restrict__ dstH,
                                                 float* __restrict__ dstF) {
  const int lane = threadIdx.x & 31;
  const int wave = threadIdx.x >> 5;
  const int hi   = lane >> 4;   // which 16-lane half
  const int ln   = lane & 15;
  const int m0 = blockIdx.x * 128 + wave * 16;  // wave's 16 output rows
  const int n0 = blockIdx.y * 64;               // block's 64 output cols

  v8f acc[4] = {};

  const _Float16* arow = A + (size_t)(m0 + ln) * C_DIM + 8 * hi;
  const _Float16* wrow = W + (size_t)(n0 + ln) * C_DIM + 8 * hi;

  // prologue: fragments for k0 = 0
  v16h af = ld_afrag(arow);
  v16h bf[4];
#pragma unroll
  for (int f = 0; f < 4; ++f)
    bf[f] = ld_afrag(wrow + (size_t)f * 16 * C_DIM);

  for (int k0 = 32; k0 < C_DIM; k0 += 32) {
    // fetch next-k fragments first (overlaps with the WMMAs below)
    v16h afn = ld_afrag(arow + k0);
    v16h bfn[4];
#pragma unroll
    for (int f = 0; f < 4; ++f)
      bfn[f] = ld_afrag(wrow + (size_t)f * 16 * C_DIM + k0);
#pragma unroll
    for (int f = 0; f < 4; ++f) acc[f] = wmma16(af, bf[f], acc[f]);
    af = afn;
#pragma unroll
    for (int f = 0; f < 4; ++f) bf[f] = bfn[f];
  }
#pragma unroll
  for (int f = 0; f < 4; ++f) acc[f] = wmma16(af, bf[f], acc[f]);

#pragma unroll
  for (int f = 0; f < 4; ++f) {
    const int n = n0 + f * 16 + ln;
    const float b = bias[n];
#pragma unroll
    for (int r = 0; r < 8; ++r) {
      const int m = m0 + r + 8 * hi;  // C/D frag: M = r + 8*(lane>>4), N = lane&15
      float v = acc[f][r] + b;
      if (MODE == 0) {
        const int h = n >> 6, d = n & 63;
        dstH[((size_t)h * N_TOK + m) * D_HEAD + d] = (_Float16)(v * ATTN_SCALE);
      } else if (MODE == 1) {
        const int h = n >> 6, d = n & 63;
        dstH[((size_t)h * N_TOK + m) * D_HEAD + d] = (_Float16)v;
      } else if (MODE == 2) {
        const int h = n >> 6, d = n & 63;
        dstH[((size_t)h * D_HEAD + d) * N_TOK + m] = (_Float16)v;  // V^T
      } else {
        dstF[(size_t)m * C_DIM + n] = v;
      }
    }
  }
}

// ---------------------------------------------------------------------------
// Flash attention, LDS-staged & double-buffered.
// Block 256 thr = 8 waves; each wave owns a 16-query tile; block = 128 queries
// of one head. All 8 waves share the K/V tiles, so tiles are staged in LDS
// (64 keys/step: Ks 8KB + Vs 8KB, x2 buffers = 32KB of the WGP's 320KB).
// Pipeline per step: issue global loads for step i+1 -> compute step i from
// LDS (8 S^T WMMAs + online softmax + 8 PV WMMAs) -> ds_store step i+1 ->
// barrier. Global latency hides under the full compute block.
// S^T = K*Q^T so the softmax reduction is per-lane + one shfl_xor(16);
// O^T = V^T*P^T with P^T taken from the S^T C-frags as-is, no lane swaps.
// ---------------------------------------------------------------------------
__global__ __launch_bounds__(256) void flash_attn(const _Float16* __restrict__ Qh,
                                                  const _Float16* __restrict__ Kh,
                                                  const _Float16* __restrict__ Vt,
                                                  _Float16* __restrict__ Xh) {
  __shared__ _Float16 Ks[2][64 * 64];  // [buf][j-local(64) * 64 d]
  __shared__ _Float16 Vs[2][64 * 64];  // [buf][d(64) * 64 j-local]

  const int tid  = threadIdx.x;
  const int lane = tid & 31;
  const int wave = tid >> 5;
  const int hi   = lane >> 4;
  const int ln   = lane & 15;
  const int h  = blockIdx.y;
  const int q0 = blockIdx.x * 128 + wave * 16;

  const _Float16* Q = Qh + (size_t)h * N_TOK * D_HEAD;
  const _Float16* K = Kh + (size_t)h * N_TOK * D_HEAD;
  const _Float16* V = Vt + (size_t)h * D_HEAD * N_TOK;

  // Per-thread cooperative-copy coordinates (16 halfs = 32B per thread/tile).
  // K tile rows are contiguous in memory: flat copy. V^T tile: 64 rows (d),
  // 4 threads per row.
  const int vRow = tid >> 2;
  const int vCol = (tid & 3) * 16;

  // Q as B fragments (N = query i = ln, K-dim = d): same split layout as A frags.
  const _Float16* qp = Q + (size_t)(q0 + ln) * D_HEAD + 8 * hi;
  const v16h qb0 = ld_afrag(qp);       // d in [0,32)
  const v16h qb1 = ld_afrag(qp + 32);  // d in [32,64)

  // preload tile 0 into buffer 0
  v16h kreg = *reinterpret_cast<const v16h*>(K + tid * 16);
  v16h vreg = *reinterpret_cast<const v16h*>(V + (size_t)vRow * N_TOK + vCol);
  *reinterpret_cast<v16h*>(&Ks[0][tid * 16]) = kreg;
  *reinterpret_cast<v16h*>(&Vs[0][vRow * 64 + vCol]) = vreg;
  __syncthreads();

  v8f o[4] = {};  // O^T: d-subtile t, (d = 16t + r + 8*hi, i = ln)
  float mrun = -3.0e38f, lrun = 0.0f;

  for (int it = 0; it < N_TOK / 64; ++it) {
    const int cur  = it & 1;
    const int j0n  = (it + 1) * 64;
    const bool more = j0n < N_TOK;

    // issue next tile's global loads; they stay in flight during compute
    if (more) {
      kreg = *reinterpret_cast<const v16h*>(K + (size_t)j0n * D_HEAD + tid * 16);
      vreg = *reinterpret_cast<const v16h*>(V + (size_t)vRow * N_TOK + j0n + vCol);
    }

    // ---- S^T: four 16-key subtiles; lane holds (j = 16*sub + r + 8*hi, i=ln)
    v8f s[4];
#pragma unroll
    for (int sub = 0; sub < 4; ++sub) {
      const _Float16* kp = &Ks[cur][(16 * sub + ln) * 64 + 8 * hi];
      v16h ka0 = ld_afrag(kp);       // d 0..31
      v16h ka1 = ld_afrag(kp + 32);  // d 32..63
      v8f c = {};
      c = wmma16(ka0, qb0, c);
      c = wmma16(ka1, qb1, c);
      s[sub] = c;
    }

    // ---- online softmax over the 64 keys of this step (per query col i=ln)
    float smax = -3.0e38f;
#pragma unroll
    for (int sub = 0; sub < 4; ++sub)
#pragma unroll
      for (int r = 0; r < 8; ++r) smax = fmaxf(smax, s[sub][r]);
    smax = fmaxf(smax, __shfl_xor(smax, 16));
    const float mnew  = fmaxf(mrun, smax);
    const float alpha = __expf(mrun - mnew);

    float lsum = 0.0f;
#pragma unroll
    for (int sub = 0; sub < 4; ++sub)
#pragma unroll
      for (int r = 0; r < 8; ++r) {
        s[sub][r] = __expf(s[sub][r] - mnew);  // p overwrites s in place
        lsum += s[sub][r];
      }
    lsum += __shfl_xor(lsum, 16);
    lrun = lrun * alpha + lsum;
    mrun = mnew;
#pragma unroll
    for (int t = 0; t < 4; ++t)
#pragma unroll
      for (int r = 0; r < 8; ++r) o[t][r] *= alpha;

    // ---- PV: two 32-key chunks. P^T B frag wants (K = j = r + 8*hi / 16 + r + 8*hi, N = ln);
    // C-frag halves hold j = 16*sub + r + 8*hi -> no lane swaps needed.
    const bool lolane = (hi == 0);
#pragma unroll
    for (int jc = 0; jc < 2; ++jc) {
      v16h pb;
#pragma unroll
      for (int r = 0; r < 8; ++r) {
        const float a  = s[2 * jc][r];
        const float b  = s[2 * jc + 1][r];
        const float ax = __shfl_xor(a, 16);
        const float bx = __shfl_xor(b, 16);
        pb[r]     = (_Float16)a;
        pb[r + 8] = (_Float16)b;
      }
#pragma unroll
      for (int t = 0; t < 4; ++t) {
        const _Float16* vp = &Vs[cur][(16 * t + ln) * 64 + 32 * jc + 8 * hi];
        v16h va = ld_afrag(vp);  // A frag: row d-local = ln, K = j-local
        o[t] = wmma16(va, pb, o[t]);
      }
    }

    // ---- write next tile into the other buffer, then block barrier
    if (more) {
      *reinterpret_cast<v16h*>(&Ks[cur ^ 1][tid * 16]) = kreg;
      *reinterpret_cast<v16h*>(&Vs[cur ^ 1][vRow * 64 + vCol]) = vreg;
    }
    __syncthreads();
  }

  const float inv = 1.0f / lrun;
#pragma unroll
  for (int t = 0; t < 4; ++t) {
    v8h outv;
#pragma unroll
    for (int r = 0; r < 8; ++r) outv[r] = (_Float16)(o[t][r] * inv);
    // row n = q0+ln, cols c = h*64 + 16t + 8*hi + [0..7]  (one 16B store)
    *reinterpret_cast<v8h*>(Xh + (size_t)(q0 + ln) * C_DIM + h * D_HEAD +
                            t * 16 + 8 * hi) = outv;
  }
}

// ---------------------------------------------------------------------------
extern "C" void kernel_launch(void* const* d_in, const int* in_sizes, int n_in,
                              void* d_out, int out_size, void* d_ws, size_t ws_size,
                              hipStream_t stream) {
  const float* feature = (const float*)d_in[0];
  const float* Wq      = (const float*)d_in[1];
  const float* bq      = (const float*)d_in[2];
  const float* Wkv     = (const float*)d_in[3];
  const float* bkv     = (const float*)d_in[4];
  const float* Wp      = (const float*)d_in[5];
  const float* bp      = (const float*)d_in[6];
  float* out = (float*)d_out;

  char* ws = (char*)d_ws;
  _Float16* featH = (_Float16*)(ws + ((size_t)0 << 20));
  _Float16* WqH   = (_Float16*)(ws + ((size_t)8 << 20));
  _Float16* WkvH  = (_Float16*)(ws + ((size_t)10 << 20));
  _Float16* WpH   = (_Float16*)(ws + ((size_t)14 << 20));
  _Float16* Qh    = (_Float16*)(ws + ((size_t)16 << 20));
  _Float16* Kh    = (_Float16*)(ws + ((size_t)24 << 20));
  _Float16* VtH   = (_Float16*)(ws + ((size_t)32 << 20));
  _Float16* Xh    = (_Float16*)(ws + ((size_t)40 << 20));

  const int nFeat = N_TOK * C_DIM;       // 4 Mi
  const int nWq   = C_DIM * C_DIM;       // 1 Mi
  const int nWkv  = 2 * C_DIM * C_DIM;   // 2 Mi
  cvt_f32_f16<<<nFeat / 2048, 256, 0, stream>>>(feature, featH, nFeat);
  cvt_f32_f16<<<nWq   / 2048, 256, 0, stream>>>(Wq,  WqH,  nWq);
  cvt_f32_f16<<<nWkv  / 2048, 256, 0, stream>>>(Wkv, WkvH, nWkv);
  cvt_f32_f16<<<nWq   / 2048, 256, 0, stream>>>(Wp,  WpH,  nWq);

  dim3 gGemm(N_TOK / 128, C_DIM / 64);  // (32, 16)
  gemm_wmma<0><<<gGemm, 256, 0, stream>>>(featH, WqH, bq, Qh, nullptr);
  gemm_wmma<1><<<gGemm, 256, 0, stream>>>(featH, WkvH, bkv, Kh, nullptr);
  gemm_wmma<2><<<gGemm, 256, 0, stream>>>(featH, WkvH + (size_t)C_DIM * C_DIM,
                                          bkv + C_DIM, VtH, nullptr);

  dim3 gAttn(N_TOK / 128, N_HEAD);      // (32, 16)
  flash_attn<<<gAttn, 256, 0, stream>>>(Qh, Kh, VtH, Xh);

  gemm_wmma<3><<<gGemm, 256, 0, stream>>>(Xh, WpH, bp, nullptr, out);
}
